// Conv2d_30691836297559
// MI455X (gfx1250) — matrix operand order repaired and checked by reading, not compiled, not run on hardware
//
#include <hip/hip_runtime.h>

typedef __attribute__((ext_vector_type(16))) __bf16 v16bf;
typedef __attribute__((ext_vector_type(8)))  float  v8f;
typedef __attribute__((ext_vector_type(4)))  int    v4i;

#define IN_C  128
#define OUT_C 256
#define Hd 64
#define Wd 64

// ---- CDNA5 async global->LDS path (probe via __has_builtin, safe fallback) ----
#if defined(__has_builtin)
#  if __has_builtin(__builtin_amdgcn_global_load_async_to_lds_b128)
#    define ASYNC_LDS 1
#  endif
#endif

typedef __attribute__((address_space(1))) v4i* gv4i_p;   // global int4*
typedef __attribute__((address_space(3))) v4i* lv4i_p;   // LDS int4*

__device__ __forceinline__ void cp16_to_lds(const unsigned short* g, unsigned short* l) {
#ifdef ASYNC_LDS
    __builtin_amdgcn_global_load_async_to_lds_b128((gv4i_p)g, (lv4i_p)l, 0, 0);
#else
    *(uint4*)l = *(const uint4*)g;
#endif
}

__device__ __forceinline__ void wait_async_lds() {
#ifdef ASYNC_LDS
#  if __has_builtin(__builtin_amdgcn_s_wait_asynccnt)
    __builtin_amdgcn_s_wait_asynccnt(0);
#  else
    asm volatile("s_wait_asynccnt 0" ::: "memory");
#  endif
#endif
}

// round-to-nearest-even f32 -> bf16 (as raw u16)
__device__ __forceinline__ unsigned f2bf_u(float f) {
    union { float f; unsigned u; } c; c.f = f;
    return (c.u + 0x7FFFu + ((c.u >> 16) & 1u)) >> 16;
}

// ---------------------------------------------------------------------------
// Pre-pass 1: NCHW f32 -> NHWC bf16 (ws_in[((b*H+y)*W+x)*128 + ic])
// ---------------------------------------------------------------------------
__global__ void __launch_bounds__(256) k_in_nhwc(const float* __restrict__ in,
                                                 unsigned short* __restrict__ ws) {
    const int by  = blockIdx.x;            // b*64 + y
    const int t   = threadIdx.x;
    const int x   = t & 63;
    const int icg = (t >> 6) << 5;         // 0,32,64,96
    const float* src = in + ((size_t)(by >> 6) * IN_C + icg) * (Hd * Wd)
                          + (size_t)(by & 63) * Wd + x;
    unsigned p[16];
#pragma unroll
    for (int i = 0; i < 16; ++i) {
        unsigned lo = f2bf_u(src[(size_t)(2 * i)     * (Hd * Wd)]);
        unsigned hi = f2bf_u(src[(size_t)(2 * i + 1) * (Hd * Wd)]);
        p[i] = lo | (hi << 16);
    }
    unsigned short* dst = ws + ((size_t)by * Wd + x) * IN_C + icg;
    uint4* d4 = (uint4*)dst;
    d4[0] = make_uint4(p[0],  p[1],  p[2],  p[3]);
    d4[1] = make_uint4(p[4],  p[5],  p[6],  p[7]);
    d4[2] = make_uint4(p[8],  p[9],  p[10], p[11]);
    d4[3] = make_uint4(p[12], p[13], p[14], p[15]);
}

// ---------------------------------------------------------------------------
// Pre-pass 2: weights OIHW f32 -> [tap][ic][oc] bf16 (oc contiguous)
// ---------------------------------------------------------------------------
__global__ void __launch_bounds__(256) k_w_tap(const float* __restrict__ w,
                                               unsigned short* __restrict__ wsw) {
    const int t = blockIdx.x * 256 + threadIdx.x;   // oc*128 + ic
    if (t >= OUT_C * IN_C) return;
    const int oc = t >> 7, ic = t & 127;
    const float* s = w + (size_t)t * 9;             // ((oc*128+ic)*3+ky)*3+kx
#pragma unroll
    for (int tap = 0; tap < 9; ++tap)
        wsw[((size_t)tap * IN_C + ic) * OUT_C + oc] = (unsigned short)f2bf_u(s[tap]);
}

// ---------------------------------------------------------------------------
// Main conv: implicit GEMM, bf16 WMMA, f32 accumulate.
// Block tile: 128 pixels (2 rows x 64 cols) x 128 oc. 8 waves: 4 M-splits x 2 N-splits.
// Wave tile: 32M x 64N = 2x4 accumulators of 16x16.
// K loop: 4 ic-chunks x 3 kernel-rows (3 taps staged per barrier pair).
// ---------------------------------------------------------------------------
#define SIN_STRIDE 40   // 32 ic + 8 pad (16B-aligned, spreads 64 LDS banks)
#define SW_STRIDE  136  // 128 oc + 8 pad

__global__ void __launch_bounds__(256) k_conv(const unsigned short* __restrict__ wsin,
                                              const unsigned short* __restrict__ wsw,
                                              const float* __restrict__ bias,
                                              float* __restrict__ out) {
    __shared__ __align__(16) unsigned short sIn[264 * SIN_STRIDE];     // 4 rows x 66 cols x 32 ic
    __shared__ __align__(16) unsigned short sW [3 * 32 * SW_STRIDE];   // 3 taps x 32 ic x 128 oc

    const int tid  = threadIdx.x;
    const int lane = tid & 31;
    const int l16  = lane & 15;
    const int half = lane >> 4;
    const int wv   = tid >> 5;
    const int wm   = (wv & 3) << 5;   // wave M offset: 0,32,64,96
    const int wn   = (wv >> 2) << 6;  // wave N offset: 0,64

    const int n0   = blockIdx.x << 7;   // oc tile base (0 or 128)
    const int p0   = blockIdx.y << 7;   // pixel tile base (128 pixels)
    const int bimg = p0 >> 12;
    const int prem = p0 & 4095;
    const int y0   = prem >> 6;         // even; tile covers image rows y0, y0+1

    v8f acc[2][4];
#pragma unroll
    for (int bt = 0; bt < 4; ++bt) {
        const float bv = bias[n0 + wn + bt * 16 + l16];   // one N per lane; all 8 C VGPRs share N
#pragma unroll
        for (int at = 0; at < 2; ++at)
            acc[at][bt] = (v8f){bv, bv, bv, bv, bv, bv, bv, bv};
    }

    for (int ic0 = 0; ic0 < IN_C; ic0 += 32) {
        __syncthreads();   // previous chunk's fragment reads of sIn complete
        // Stage input slab (async): rows y0-1..y0+2, cols -1..64, 32 ch; zero-fill halo.
        for (int site = tid; site < 264; site += 256) {
            const int r  = site / 66;
            const int cc = site - r * 66;
            const int yy = y0 - 1 + r;
            const int xx = cc - 1;
            unsigned short* l = sIn + site * SIN_STRIDE;
            if ((unsigned)yy < Hd && (unsigned)xx < Wd) {
                const unsigned short* g = wsin +
                    (((size_t)bimg * Hd + yy) * Wd + xx) * IN_C + ic0;
                cp16_to_lds(g,      l);
                cp16_to_lds(g + 8,  l + 8);
                cp16_to_lds(g + 16, l + 16);
                cp16_to_lds(g + 24, l + 24);
            } else {
                const uint4 z = make_uint4(0, 0, 0, 0);
                uint4* d = (uint4*)l;
                d[0] = z; d[1] = z; d[2] = z; d[3] = z;
            }
        }
        for (int ky = 0; ky < 3; ++ky) {
            __syncthreads();   // previous row's fragment reads of sW complete
            // Stage 3 weight tiles (taps ky*3+0..2), each 32 ic x 128 oc (async)
            for (int q = tid; q < 1536; q += 256) {
                const int kx  = q >> 9;
                const int r   = q & 511;
                const int icr = r >> 4;
                const int ocq = r & 15;
                cp16_to_lds(wsw + ((size_t)(ky * 3 + kx) * IN_C + ic0 + icr) * OUT_C
                                + n0 + ocq * 8,
                            sW + (kx * 32 + icr) * SW_STRIDE + ocq * 8);
            }
            wait_async_lds();   // this thread's async transfers (incl. slab) done
            __syncthreads();    // all threads' staging visible

            for (int kx = 0; kx < 3; ++kx) {
                union Frag { uint4 u[2]; v16bf v; };
                Frag fa[2], fb[4];
                // A fragments: lane -> M; K halves at {half*8, 16+half*8} (ISA 16-bit A layout)
#pragma unroll
                for (int at = 0; at < 2; ++at) {
                    const int m  = wm + at * 16 + l16;    // pixel within 128-tile
                    const int r2 = (m >> 6) + ky;         // slab row 0..3
                    const int cc = (m & 63) + kx;         // slab col 0..65
                    const unsigned short* ap = sIn + (r2 * 66 + cc) * SIN_STRIDE + half * 8;
                    fa[at].u[0] = *(const uint4*)ap;
                    fa[at].u[1] = *(const uint4*)(ap + 16);
                }
                // B fragments: lane -> N; K halves at {half*8, 16+half*8} (same as A)
#pragma unroll
                for (int bt = 0; bt < 4; ++bt) {
                    const unsigned short* bp = sW + (kx * 32 + half * 8) * SW_STRIDE
                                             + wn + bt * 16 + l16;
                    const unsigned short* bq = bp + 16 * SW_STRIDE;
                    fb[bt].u[0] = make_uint4(
                        (unsigned)bp[0] | ((unsigned)bp[SW_STRIDE] << 16),
                        (unsigned)bp[2 * SW_STRIDE] | ((unsigned)bp[3 * SW_STRIDE] << 16),
                        (unsigned)bp[4 * SW_STRIDE] | ((unsigned)bp[5 * SW_STRIDE] << 16),
                        (unsigned)bp[6 * SW_STRIDE] | ((unsigned)bp[7 * SW_STRIDE] << 16));
                    fb[bt].u[1] = make_uint4(
                        (unsigned)bq[0] | ((unsigned)bq[SW_STRIDE] << 16),
                        (unsigned)bq[2 * SW_STRIDE] | ((unsigned)bq[3 * SW_STRIDE] << 16),
                        (unsigned)bq[4 * SW_STRIDE] | ((unsigned)bq[5 * SW_STRIDE] << 16),
                        (unsigned)bq[6 * SW_STRIDE] | ((unsigned)bq[7 * SW_STRIDE] << 16));
                }
#pragma unroll
                for (int at = 0; at < 2; ++at)
#pragma unroll
                    for (int bt = 0; bt < 4; ++bt)
                        acc[at][bt] = __builtin_amdgcn_wmma_f32_16x16x32_bf16(
                            false, fa[at].v, false, fb[bt].v,
                            (short)0, acc[at][bt], false, false);
            }
        }
    }

    // Store: C layout VGPR v -> M = v + 8*half, N = l16. v..v+3 = consecutive x => b128 stores.
#pragma unroll
    for (int at = 0; at < 2; ++at) {
#pragma unroll
        for (int bt = 0; bt < 4; ++bt) {
            const int oc  = n0 + wn + bt * 16 + l16;
            const int pix = prem + wm + at * 16 + half * 8;
            float* o = out + (((size_t)bimg * OUT_C + oc) << 12) + pix;
            const float4* cv = (const float4*)&acc[at][bt];
            *(float4*)o       = cv[0];
            *(float4*)(o + 4) = cv[1];
        }
    }
}

// ---------------------------------------------------------------------------
extern "C" void kernel_launch(void* const* d_in, const int* in_sizes, int n_in,
                              void* d_out, int out_size, void* d_ws, size_t ws_size,
                              hipStream_t stream) {
    const float* x  = (const float*)d_in[0];   // [32,128,64,64] f32
    const float* wt = (const float*)d_in[1];   // [256,128,3,3] f32
    const float* bs = (const float*)d_in[2];   // [256] f32
    float* out = (float*)d_out;                // [32,256,64,64] f32

    const size_t ws_in_bytes = (size_t)32 * Hd * Wd * IN_C * sizeof(unsigned short); // 33.5 MB
    unsigned short* ws_in = (unsigned short*)d_ws;
    unsigned short* ws_w  = (unsigned short*)((char*)d_ws + ws_in_bytes);            // +576 KB

    k_in_nhwc<<<32 * Hd, 256, 0, stream>>>(x, ws_in);
    k_w_tap<<<(OUT_C * IN_C + 255) / 256, 256, 0, stream>>>(wt, ws_w);

    dim3 grid(OUT_C / 128, (32 * Hd * Wd) / 128);  // (2, 1024)
    k_conv<<<grid, 256, 0, stream>>>(ws_in, ws_w, bs, out);
}
